// LinearAttention_7962869367147
// MI455X (gfx1250) — matrix operand order repaired and checked by reading, not compiled, not run on hardware
//
#include <hip/hip_runtime.h>

// ---------------------------------------------------------------------------
// Causal linear attention (chunked), MI455X / gfx1250, wave32 + WMMA f16->f32
// All WMMA operands are stored in LDS so fragment reads are contiguous and
// vectorized (ds_load_b128), B-side matrices kept transposed in LDS.
// ---------------------------------------------------------------------------

typedef __attribute__((ext_vector_type(16))) _Float16 v16h;
typedef __attribute__((ext_vector_type(8)))  _Float16 v8h;
typedef __attribute__((ext_vector_type(4)))  _Float16 v4h;
typedef __attribute__((ext_vector_type(8)))  float    v8f;

#define B_    4
#define L_    2048
#define H_    12
#define D_    64
#define C_    128
#define G_    16           // L_/C_
#define NHB   (B_*H_)      // 48
#define NBLK  (B_*H_*G_)   // 768
#define KVSZ  (D_*D_)      // 4096
#define LDA   72           // padded stride (halves) for 64-wide arrays (mult of 8)
#define LDT   136          // padded stride (halves) for 128-wide arrays (mult of 8)

__device__ __forceinline__ float phi_f(float x) {
    // elu(x)+1  (strictly positive feature map)
    return x > 0.0f ? x + 1.0f : __expf(x);
}

__device__ __forceinline__ v8f wmma16(v16h a, v16h b, v8f c) {
    return __builtin_amdgcn_wmma_f32_16x16x32_f16(
        /*neg_a=*/false, a, /*neg_b=*/false, b,
        /*c_mod=*/(short)0, c, /*reuse_a=*/false, /*reuse_b=*/false);
}

__device__ __forceinline__ v16h join8(v8h lo, v8h hi) {
    return __builtin_shufflevector(lo, hi, 0, 1, 2, 3, 4, 5, 6, 7,
                                            8, 9, 10, 11, 12, 13, 14, 15);
}

// A fragment (16x32 f16): A row-major [M x K] in LDS, stride ld (halves, mult 8).
// Lane = 16*half + m. Elements e<8 -> K = k0+8*half+e ; e>=8 -> K = k0+16+8*half+(e-8).
// Both groups are 8 contiguous halves (16 B, aligned) -> two ds_load_b128.
__device__ __forceinline__ v16h load_a(const _Float16* A, int ld, int row0, int k0, int lane) {
    const int half = lane >> 4, m = lane & 15;
    const _Float16* p = A + (row0 + m) * ld + k0 + 8 * half;
    const v8h lo = *(const v8h*)(p);
    const v8h hi = *(const v8h*)(p + 16);
    return join8(lo, hi);
}

// B fragment (32x16 f16) from TRANSPOSED storage: Bt row-major with Bt[n][k] = B[k][n].
// Lane = 16*half + n; e<8 -> K = k0+8*half+e ; e>=8 -> K = k0+16+8*half+(e-8).
__device__ __forceinline__ v16h load_bT(const _Float16* Bt, int ld, int k0, int col0, int lane) {
    const int half = lane >> 4, n = lane & 15;
    const _Float16* p = Bt + (col0 + n) * ld + k0 + 8 * half;
    const v8h lo = *(const v8h*)(p);
    const v8h hi = *(const v8h*)(p + 16);
    return join8(lo, hi);
}

// ---------------------------------------------------------------------------
// Pass 1: per-chunk  kv[n,h,g] = k_phi^T @ v   (64x64)  and  ksum[n,h,g][d]
// ---------------------------------------------------------------------------
__global__ void la_pass1(const float* __restrict__ k_g, const float* __restrict__ v_g,
                         float* __restrict__ kv_ws, float* __restrict__ ksum_ws) {
    __shared__ __align__(16) _Float16 ksT[D_ * LDT];   // ksT[d][c] = phi(k)[c][d]
    __shared__ __align__(16) _Float16 vsT[D_ * LDT];   // vsT[d][c] = v[c][d]

    const int blk = blockIdx.x;            // (n*H + h)*G + g
    const int g = blk % G_;
    const int h = (blk / G_) % H_;
    const int n = blk / (G_ * H_);
    const int t = threadIdx.x;             // 256 threads = 8 waves

    for (int i = t * 4; i < C_ * D_; i += 1024) {
        const int c = i >> 6, d = i & 63;
        const int gl = ((n * L_ + g * C_ + c) * H_ + h) * D_ + d;
        const float4 kq = *(const float4*)(k_g + gl);
        const float4 vq = *(const float4*)(v_g + gl);
        ksT[(d + 0) * LDT + c] = (_Float16)phi_f(kq.x);
        ksT[(d + 1) * LDT + c] = (_Float16)phi_f(kq.y);
        ksT[(d + 2) * LDT + c] = (_Float16)phi_f(kq.z);
        ksT[(d + 3) * LDT + c] = (_Float16)phi_f(kq.w);
        vsT[(d + 0) * LDT + c] = (_Float16)vq.x;
        vsT[(d + 1) * LDT + c] = (_Float16)vq.y;
        vsT[(d + 2) * LDT + c] = (_Float16)vq.z;
        vsT[(d + 3) * LDT + c] = (_Float16)vq.w;
    }
    __syncthreads();

    const int wave = t >> 5, lane = t & 31;
    const int mbase = 8 * (lane >> 4), nn = lane & 15;
    float* kvout = kv_ws + (size_t)blk * KVSZ;

    // dS = ks^T @ vs : A = ksT (row-major [64 x 128]), B = vs via vsT.
    // 4x4 tiles of 16x16, 2 tiles per wave, K = 128 (4 steps)
#pragma unroll
    for (int ti = wave * 2; ti < wave * 2 + 2; ++ti) {
        const int row0 = (ti >> 2) * 16;   // d1
        const int col0 = (ti & 3) * 16;    // d2
        v8f acc = {};
#pragma unroll
        for (int kk = 0; kk < C_; kk += 32) {
            v16h a = load_a (ksT, LDT, row0, kk, lane);
            v16h b = load_bT(vsT, LDT, kk, col0, lane);
            acc = wmma16(a, b, acc);
        }
#pragma unroll
        for (int r = 0; r < 8; ++r)
            kvout[(row0 + mbase + r) * D_ + col0 + nn] = acc[r];
    }

    // column sums of k_phi (contiguous rows of ksT)
    if (t < D_) {
        float s = 0.0f;
        for (int c = 0; c < C_; ++c) s += (float)ksT[t * LDT + c];
        ksum_ws[(size_t)blk * D_ + t] = s;
    }
}

// ---------------------------------------------------------------------------
// Pass 2: exclusive prefix-scan over chunks g for kv and ksum (in place)
// ---------------------------------------------------------------------------
__global__ void la_pass2(float* __restrict__ kv_ws, float* __restrict__ ksum_ws) {
    const int nh = blockIdx.x;             // 0..47
    const int t = threadIdx.x;             // 256

    float run[16];
#pragma unroll
    for (int j = 0; j < 16; ++j) run[j] = 0.0f;

    for (int g = 0; g < G_; ++g) {
        float* p = kv_ws + ((size_t)nh * G_ + g) * KVSZ;
#pragma unroll
        for (int j = 0; j < 16; ++j) {
            const int e = j * 256 + t;     // coalesced; each (j,t) owns element e
            const float tmp = p[e];
            p[e] = run[j];
            run[j] += tmp;
        }
    }
    if (t < D_) {
        float r = 0.0f;
        for (int g = 0; g < G_; ++g) {
            float* p = ksum_ws + ((size_t)nh * G_ + g) * D_;
            const float tmp = p[t];
            p[t] = r;
            r += tmp;
        }
    }
}

// ---------------------------------------------------------------------------
// Pass 3: per chunk: scores -> mask -> z ; out = (mask(scores)@v + q@S_excl)/z
// ---------------------------------------------------------------------------
__global__ void la_pass3(const float* __restrict__ q_g, const float* __restrict__ k_g,
                         const float* __restrict__ v_g,
                         const float* __restrict__ kv_ws, const float* __restrict__ ksum_ws,
                         float* __restrict__ out) {
    __shared__ __align__(16) _Float16 qs [C_ * LDA];   // 18 KB, row-major (A operand)
    __shared__ __align__(16) _Float16 ks [C_ * LDA];   // 18 KB, row-major (B^T operand)
    __shared__ __align__(16) _Float16 vsT[D_ * LDT];   // 17 KB, vsT[d][c] (B operand)
    __shared__ __align__(16) _Float16 ss [C_ * LDT];   // 34 KB, masked scores (A operand)
    __shared__ __align__(16) _Float16 SfT[D_ * LDA];   //  9 KB, SfT[d2][d1] (B operand)
    __shared__ float zrow[C_];
    __shared__ float carry[D_];

    const int blk = blockIdx.x;
    const int g = blk % G_;
    const int h = (blk / G_) % H_;
    const int n = blk / (G_ * H_);
    const int t = threadIdx.x;

    for (int i = t * 4; i < C_ * D_; i += 1024) {
        const int c = i >> 6, d = i & 63;
        const int gl = ((n * L_ + g * C_ + c) * H_ + h) * D_ + d;
        const float4 qq = *(const float4*)(q_g + gl);
        const float4 kq = *(const float4*)(k_g + gl);
        const float4 vq = *(const float4*)(v_g + gl);
        const v4h qh = { (_Float16)phi_f(qq.x), (_Float16)phi_f(qq.y),
                         (_Float16)phi_f(qq.z), (_Float16)phi_f(qq.w) };
        const v4h kh = { (_Float16)phi_f(kq.x), (_Float16)phi_f(kq.y),
                         (_Float16)phi_f(kq.z), (_Float16)phi_f(kq.w) };
        *(v4h*)(qs + c * LDA + d) = qh;    // 8-byte ds_store_b64
        *(v4h*)(ks + c * LDA + d) = kh;
        vsT[(d + 0) * LDT + c] = (_Float16)vq.x;
        vsT[(d + 1) * LDT + c] = (_Float16)vq.y;
        vsT[(d + 2) * LDT + c] = (_Float16)vq.z;
        vsT[(d + 3) * LDT + c] = (_Float16)vq.w;
    }
    for (int i = t; i < KVSZ; i += 256) {
        const int d1 = i >> 6, d2 = i & 63;
        SfT[d2 * LDA + d1] = (_Float16)kv_ws[(size_t)blk * KVSZ + i];
    }
    if (t < D_) carry[t] = ksum_ws[(size_t)blk * D_ + t];
    __syncthreads();

    const int wave = t >> 5, lane = t & 31;
    const int mbase = 8 * (lane >> 4), nn = lane & 15;

    // scores = q_phi @ k_phi^T : B^T = ks rows (contiguous). 8x8 tiles, 8/wave.
#pragma unroll
    for (int ti = wave * 8; ti < wave * 8 + 8; ++ti) {
        const int row0 = (ti >> 3) * 16, col0 = (ti & 7) * 16;
        v8f acc = {};
#pragma unroll
        for (int kk = 0; kk < D_; kk += 32) {
            v16h a = load_a (qs, LDA, row0, kk, lane);
            v16h b = load_bT(ks, LDA, kk, col0, lane);   // B[k][n] = ks[n][k]
            acc = wmma16(a, b, acc);
        }
#pragma unroll
        for (int r = 0; r < 8; ++r) {
            const int row = row0 + mbase + r, col = col0 + nn;
            const float val = (col <= row) ? acc[r] : 0.0f;   // causal (s <= c)
            ss[row * LDT + col] = (_Float16)val;
        }
    }
    __syncthreads();

    // z[c] = eps + q_phi[c].carry + sum_{s<=c} scores[c][s]
    if (t < C_) {
        float s = 1e-6f;
        for (int d = 0; d < D_; ++d) s += (float)qs[t * LDA + d] * carry[d];
        for (int c = 0; c <= t; ++c) s += (float)ss[t * LDT + c];
        zrow[t] = s;
    }
    __syncthreads();

    // out = (mask(scores)@v + q_phi@S_excl) / z : 8x4 tiles, 4 per wave
#pragma unroll
    for (int ti = wave * 4; ti < wave * 4 + 4; ++ti) {
        const int row0 = (ti >> 2) * 16, col0 = (ti & 3) * 16;
        v8f acc = {};
#pragma unroll
        for (int kk = 0; kk < C_; kk += 32) {           // intra: K = 128
            v16h a = load_a (ss, LDT, row0, kk, lane);
            v16h b = load_bT(vsT, LDT, kk, col0, lane);
            acc = wmma16(a, b, acc);
        }
#pragma unroll
        for (int kk = 0; kk < D_; kk += 32) {           // inter: K = 64
            v16h a = load_a (qs, LDA, row0, kk, lane);
            v16h b = load_bT(SfT, LDA, kk, col0, lane);
            acc = wmma16(a, b, acc);
        }
#pragma unroll
        for (int r = 0; r < 8; ++r) {
            const int row = row0 + mbase + r, col = col0 + nn;
            const int gl = ((n * L_ + g * C_ + row) * H_ + h) * D_ + col;
            out[gl] = acc[r] / zrow[row];
        }
    }
}

// ---------------------------------------------------------------------------
extern "C" void kernel_launch(void* const* d_in, const int* in_sizes, int n_in,
                              void* d_out, int out_size, void* d_ws, size_t ws_size,
                              hipStream_t stream) {
    const float* q = (const float*)d_in[0];
    const float* k = (const float*)d_in[1];
    const float* v = (const float*)d_in[2];
    float* out = (float*)d_out;

    // workspace layout: kv[768][64*64] f32, then ksum[768][64] f32  (~12.8 MB)
    float* kv_ws   = (float*)d_ws;
    float* ksum_ws = kv_ws + (size_t)NBLK * KVSZ;

    la_pass1<<<dim3(NBLK), dim3(256), 0, stream>>>(k, v, kv_ws, ksum_ws);
    la_pass2<<<dim3(NHB),  dim3(256), 0, stream>>>(kv_ws, ksum_ws);
    la_pass3<<<dim3(NBLK), dim3(256), 0, stream>>>(q, k, v, kv_ws, ksum_ws, out);
}
